// BatchGCN_60765197304357
// MI455X (gfx1250) — compile-verified, matrix operand order checked by reading, not run on hardware
//
#include <hip/hip_runtime.h>
#include <hip/hip_bf16.h>

typedef __attribute__((ext_vector_type(16))) _Float16 v16h;
typedef __attribute__((ext_vector_type(8)))  float    v8f;

#define B_  8
#define N_  10000
#define E_  160000
#define F_  64
#define BN_ (B_*N_)       // 80000 nodes total
#define BE_ (B_*E_)       // 1280000 edges total

// f16 B-fragment pack: [ntile(4)][kb(2)][lane(32)][16 halves] = 4096 halves = 8KB
#define WPACK_HALVES 4096

// ---------------------------------------------------------------- utilities
__global__ void zero_f32(float* __restrict__ p, int n) {
  int i = blockIdx.x * blockDim.x + threadIdx.x;
  int stride = gridDim.x * blockDim.x;
  for (; i < n; i += stride) p[i] = 0.0f;
}

// edge_index layout [B,E,2] int32; build merged row/col ids + degree
__global__ void degree_kernel(const int* __restrict__ ei, float* __restrict__ deg,
                              int* __restrict__ row32, int* __restrict__ col32) {
  int g = blockIdx.x * blockDim.x + threadIdx.x;
  if (g >= BE_) return;
  int b = g / E_;
  int r = ei[2 * g + 0] + b * N_;
  int c = ei[2 * g + 1] + b * N_;
  row32[g] = r;
  col32[g] = c;
  atomicAdd(&deg[r], 1.0f);
}

__global__ void dinv_kernel(float* __restrict__ deg) {
  int i = blockIdx.x * blockDim.x + threadIdx.x;
  if (i >= BN_) return;
  float d = deg[i];
  deg[i] = (d > 0.0f) ? rsqrtf(d) : 0.0f;
}

__global__ void norm_kernel(const float* __restrict__ dinv, const int* __restrict__ row32,
                            const int* __restrict__ col32, float* __restrict__ norm) {
  int g = blockIdx.x * blockDim.x + threadIdx.x;
  if (g >= BE_) return;
  norm[g] = dinv[row32[g]] * dinv[col32[g]];
}

// ------------------------------------------------- W -> f16 B-fragment pack
// B (math) = W^T, i.e. B[k][n] = W[n][k].  CDNA5 16-bit 16x32 fragment striping:
//   lane L: n-col = L%16, half = L/16; VGPR v: k = kb*32 + (v<4 ? 2v : 16+2(v-4)) + half*8
// One 256-thread block packs one layer (thread t -> one lane-fragment of 16 halves).
__global__ __launch_bounds__(256) void pack_w_kernel(const float* __restrict__ W,
                                                     _Float16* __restrict__ Wp) {
  const int t    = threadIdx.x;      // 0..255
  const int nt   = t >> 6;           // 0..3
  const int kb   = (t >> 5) & 1;     // 0..1
  const int lane = t & 31;
  const int n    = nt * 16 + (lane & 15);
  const int half = lane >> 4;
  _Float16* dst = Wp + ((size_t)((nt * 2 + kb) * 32 + lane)) * 16;
#pragma unroll
  for (int v = 0; v < 8; ++v) {
    int kbase = (v < 4) ? (2 * v) : (16 + 2 * (v - 4));
    int k = kb * 32 + kbase + half * 8;
    dst[2 * v]     = (_Float16)W[n * F_ + k];
    dst[2 * v + 1] = (_Float16)W[n * F_ + k + 1];
  }
}

// ------------------------------------------------- GEMM (H @ W^T + b) via WMMA
// One wave computes a 16-row x 64-col tile. K=64 split into two K=32 WMMAs.
// B fragments are pre-packed f16 (contiguous 32B per lane -> 2x b128 loads).
template <bool RELU_IN>
__global__ __launch_bounds__(256) void gemm_bias_wmma(const float* __restrict__ Hin,
                                                      const _Float16* __restrict__ Wp,
                                                      const float* __restrict__ bias,
                                                      float* __restrict__ Hout) {
  const int wave = threadIdx.x >> 5;
  const int lane = threadIdx.x & 31;
  const int rb   = blockIdx.x * 8 + wave;   // 16-row tile index (wave-uniform)
  if (rb * 16 >= BN_) return;               // uniform exit: EXEC stays all-1s
  const int row0 = rb * 16;
  const int half = lane >> 4;
  const int mrow = lane & 15;

  // ---- A fragments (two K=32 chunks), ReLU fused on input when requested
  v16h a0, a1;
  const float* hrow = Hin + (size_t)(row0 + mrow) * F_;
#pragma unroll
  for (int v = 0; v < 8; ++v) {
    int kbase = (v < 4) ? (2 * v) : (16 + 2 * (v - 4));
    int k0 = kbase + half * 8;
    float2 p0 = *(const float2*)(hrow + k0);
    float2 p1 = *(const float2*)(hrow + 32 + k0);
    if (RELU_IN) {
      p0.x = fmaxf(p0.x, 0.0f); p0.y = fmaxf(p0.y, 0.0f);
      p1.x = fmaxf(p1.x, 0.0f); p1.y = fmaxf(p1.y, 0.0f);
    }
    a0[2 * v] = (_Float16)p0.x; a0[2 * v + 1] = (_Float16)p0.y;
    a1[2 * v] = (_Float16)p1.x; a1[2 * v + 1] = (_Float16)p1.y;
  }

  const v16h* Wpv = (const v16h*)Wp;   // 32B-aligned lane fragments
#pragma unroll
  for (int nt = 0; nt < 4; ++nt) {
    const v16h b0 = Wpv[(nt * 2 + 0) * 32 + lane];
    const v16h b1 = Wpv[(nt * 2 + 1) * 32 + lane];

    v8f c = {};
    c = __builtin_amdgcn_wmma_f32_16x16x32_f16(false, a0, false, b0, (short)0, c, false, false);
    c = __builtin_amdgcn_wmma_f32_16x16x32_f16(false, a1, false, b1, (short)0, c, false, false);

    const int n = nt * 16 + mrow;
    const float bv = bias[n];
#pragma unroll
    for (int r = 0; r < 8; ++r) {
      int m = r + half * 8;                  // C/D layout: lanes16-31 hold M+8
      Hout[(size_t)(row0 + m) * F_ + n] = c[r] + bv;
    }
  }
}

// ------------------------------------------------- edge scatter (segment_sum)
// One wave per edge; each lane handles 2 consecutive features (float2 load).
__global__ __launch_bounds__(256) void aggregate_kernel(const float* __restrict__ Hlin,
                                                        const int* __restrict__ row32,
                                                        const int* __restrict__ col32,
                                                        const float* __restrict__ norm,
                                                        float* __restrict__ out) {
  const int wave = threadIdx.x >> 5;
  const int lane = threadIdx.x & 31;
  const int g = blockIdx.x * 8 + wave;
  if (g >= BE_) return;
  const int r = row32[g];
  const int c = col32[g];
  const float nv = norm[g];
  const float2 h = ((const float2*)(Hlin + (size_t)c * F_))[lane];
  float* dst = out + (size_t)r * F_ + 2 * lane;
  atomicAdd(dst + 0, nv * h.x);
  atomicAdd(dst + 1, nv * h.y);
}

// ---------------------------------------------------------------- launcher
extern "C" void kernel_launch(void* const* d_in, const int* in_sizes, int n_in,
                              void* d_out, int out_size, void* d_ws, size_t ws_size,
                              hipStream_t stream) {
  const float* x  = (const float*)d_in[0];
  const int*   ei = (const int*)  d_in[1];
  const float* W1 = (const float*)d_in[2];
  const float* b1 = (const float*)d_in[3];
  const float* W2 = (const float*)d_in[4];
  const float* b2 = (const float*)d_in[5];
  const float* W3 = (const float*)d_in[6];
  const float* b3 = (const float*)d_in[7];
  float* out = (float*)d_out;

  auto align256 = [](size_t v) { return (v + 255) & ~(size_t)255; };
  char* ws = (char*)d_ws;
  float* deg   = (float*)ws;     ws += align256((size_t)BN_ * 4);   // deg -> dinv in place
  int*   row32 = (int*)ws;       ws += align256((size_t)BE_ * 4);
  int*   col32 = (int*)ws;       ws += align256((size_t)BE_ * 4);
  float* norm  = (float*)ws;     ws += align256((size_t)BE_ * 4);
  float* Hlin  = (float*)ws;     ws += align256((size_t)BN_ * F_ * 4);
  float* Hagg  = (float*)ws;     ws += align256((size_t)BN_ * F_ * 4);
  _Float16* Wp1 = (_Float16*)ws; ws += align256((size_t)WPACK_HALVES * 2);
  _Float16* Wp2 = (_Float16*)ws; ws += align256((size_t)WPACK_HALVES * 2);
  _Float16* Wp3 = (_Float16*)ws; ws += align256((size_t)WPACK_HALVES * 2);

  const int TB = 256;
  // --- graph preprocessing + weight packing
  zero_f32<<<512, TB, 0, stream>>>(deg, BN_);
  degree_kernel<<<(BE_ + TB - 1) / TB, TB, 0, stream>>>(ei, deg, row32, col32);
  dinv_kernel<<<(BN_ + TB - 1) / TB, TB, 0, stream>>>(deg);
  norm_kernel<<<(BE_ + TB - 1) / TB, TB, 0, stream>>>(deg, row32, col32, norm);
  pack_w_kernel<<<1, TB, 0, stream>>>(W1, Wp1);
  pack_w_kernel<<<1, TB, 0, stream>>>(W2, Wp2);
  pack_w_kernel<<<1, TB, 0, stream>>>(W3, Wp3);

  const int GEMM_GRID = (BN_ / 16 + 7) / 8;   // 625
  const int AGG_GRID  = BE_ / 8;              // 160000

  // --- layer 1: h1 = relu(scatter(norm * (x@W1^T+b1)))  (relu fused into next gemm)
  gemm_bias_wmma<false><<<GEMM_GRID, TB, 0, stream>>>(x, Wp1, b1, Hlin);
  zero_f32<<<2048, TB, 0, stream>>>(Hagg, BN_ * F_);
  aggregate_kernel<<<AGG_GRID, TB, 0, stream>>>(Hlin, row32, col32, norm, Hagg);

  // --- layer 2
  gemm_bias_wmma<true><<<GEMM_GRID, TB, 0, stream>>>(Hagg, Wp2, b2, Hlin);
  zero_f32<<<2048, TB, 0, stream>>>(Hagg, BN_ * F_);
  aggregate_kernel<<<AGG_GRID, TB, 0, stream>>>(Hlin, row32, col32, norm, Hagg);

  // --- layer 3 (no relu on output), aggregate straight into d_out
  gemm_bias_wmma<true><<<GEMM_GRID, TB, 0, stream>>>(Hagg, Wp3, b3, Hlin);
  zero_f32<<<2048, TB, 0, stream>>>(out, BN_ * F_);
  aggregate_kernel<<<AGG_GRID, TB, 0, stream>>>(Hlin, row32, col32, norm, out);
}
